// EdgePooling_10316511445662
// MI455X (gfx1250) — matrix operand order repaired and checked by reading, not compiled, not run on hardware
//
#include <hip/hip_runtime.h>
#include <stdint.h>

typedef unsigned long long u64;
typedef unsigned int u32;
typedef __attribute__((ext_vector_type(16))) _Float16 v16h;
typedef __attribute__((ext_vector_type(8)))  float    v8f;

#define TPB 256

// ============================================================
// 1. Node scores via WMMA: score = sigmoid(x @ w + b)
//    One wave computes 16 rows. A = 16x32 f16 tile of x,
//    B = 32x16 f16 with w replicated in every column, so the
//    f32 accumulator holds the row dot-product in every lane.
// ============================================================
__global__ void ep_score_wmma(const float* __restrict__ x, const float* __restrict__ w,
                              const float* __restrict__ bias,
                              float* __restrict__ score_out, u64* __restrict__ nodekeys,
                              int n, int d)
{
    int gwave = (blockIdx.x * blockDim.x + threadIdx.x) >> 5;
    int lane  = threadIdx.x & 31;
    int row0  = gwave * 16;
    if (row0 >= n) return;                    // wave-uniform exit (EXEC all-1 for WMMA)
    int  m   = lane & 15;
    bool hi  = lane >= 16;
    int  row = row0 + m; if (row >= n) row = n - 1;
    float bz = bias[0];
    v8f c = {};
    for (int kb = 0; kb < d; kb += 32) {
        v16h a, b;
#pragma unroll
        for (int j = 0; j < 16; ++j) {
            // 16-bit A-matrix 16x32 layout (ISA 7.12.2)
            int ka = (j < 8) ? (j + (hi ? 8 : 0)) : (j + 8 + (hi ? 8 : 0));
            a[j] = (_Float16)x[(size_t)row * d + kb + ka];
            // B: same K order as A (table layout); all N columns = w
            b[j] = (_Float16)w[kb + ka];
        }
        c = __builtin_amdgcn_wmma_f32_16x16x32_f16(false, a, false, b, (short)0, c, false, false);
    }
    if (m == 0) {            // lanes 0 (M=0..7) and 16 (M=8..15) write results
#pragma unroll
        for (int r = 0; r < 8; ++r) {
            int rr = row0 + r + (hi ? 8 : 0);
            if (rr < n) {
                float z = c[r] + bz;
                float s = 1.0f / (1.0f + __expf(-z));
                score_out[rr] = s;
                u32 u = __float_as_uint(s);
                u = (u & 0x80000000u) ? ~u : (u | 0x80000000u);  // order-preserving key
                nodekeys[rr] = ((u64)u << 32) | (u32)rr;          // stable: tie-break by index
            }
        }
    }
}

// ============================================================
// Generic helpers
// ============================================================
__global__ void ep_pad_u64(u64* a, int from, unsigned to)
{
    unsigned p = blockIdx.x * blockDim.x + threadIdx.x + (unsigned)from;
    if (p < to) a[p] = ~0ull;
}

// global bitonic phase (j >= 1024: partner crosses 1024-chunk boundary)
__global__ void ep_bitonic(u64* a, unsigned npow, unsigned j, unsigned k)
{
    unsigned i = blockIdx.x * blockDim.x + threadIdx.x;
    if (i >= npow) return;
    unsigned ixj = i ^ j;
    if (ixj <= i) return;
    u64 ai = a[i], aj = a[ixj];
    bool up = ((i & k) == 0);
    bool sw = up ? (ai > aj) : (ai < aj);
    if (sw) { a[i] = aj; a[ixj] = ai; }
}

// LDS-fused bitonic: full local sort k=2..1024 (all phases j<=512 stay in-chunk)
__global__ void ep_bitonic_local(u64* a)
{
    __shared__ u64 sh[1024];
    unsigned t    = threadIdx.x;               // 512 threads
    unsigned base = blockIdx.x * 1024u;
    sh[t]       = a[base + t];
    sh[t + 512] = a[base + t + 512];
    __syncthreads();
    for (unsigned k = 2; k <= 1024; k <<= 1) {
        for (unsigned j = k >> 1; j >= 1; j >>= 1) {
            unsigned i = ((t & ~(j - 1)) << 1) | (t & (j - 1));
            bool up = (((base + i) & k) == 0);
            u64 ai = sh[i], aj = sh[i + j];
            bool sw = up ? (ai > aj) : (ai < aj);
            if (sw) { sh[i] = aj; sh[i + j] = ai; }
            __syncthreads();
        }
    }
    a[base + t]       = sh[t];
    a[base + t + 512] = sh[t + 512];
}

// LDS-fused bitonic: tail phases j=512..1 of merge step k (k >= 2048)
__global__ void ep_bitonic_fused(u64* a, unsigned k)
{
    __shared__ u64 sh[1024];
    unsigned t    = threadIdx.x;               // 512 threads
    unsigned base = blockIdx.x * 1024u;
    sh[t]       = a[base + t];
    sh[t + 512] = a[base + t + 512];
    __syncthreads();
    for (unsigned j = 512; j >= 1; j >>= 1) {
        unsigned i = ((t & ~(j - 1)) << 1) | (t & (j - 1));
        bool up = (((base + i) & k) == 0);
        u64 ai = sh[i], aj = sh[i + j];
        bool sw = up ? (ai > aj) : (ai < aj);
        if (sw) { sh[i] = aj; sh[i + j] = ai; }
        __syncthreads();
    }
    a[base + t]       = sh[t];
    a[base + t + 512] = sh[t + 512];
}

__global__ void ep_zero_f32(float* a, long long count)
{
    long long t = (long long)blockIdx.x * blockDim.x + threadIdx.x;
    if (t < count) a[t] = 0.0f;
}
__global__ void ep_zero_u32(u32* a, int count)
{
    int t = blockIdx.x * blockDim.x + threadIdx.x;
    if (t < count) a[t] = 0u;
}

// single-block exclusive scan over u32 (1024 threads, chunked Hillis-Steele)
__global__ void ep_scan_u32(const u32* in, u32* out, int n, u32* total_out, float* total_f_out)
{
    __shared__ u32 sh[1024];
    __shared__ u32 carry;
    int t = threadIdx.x;
    if (t == 0) carry = 0;
    __syncthreads();
    for (int base = 0; base < n; base += 1024) {
        int i = base + t;
        u32 v = (i < n) ? in[i] : 0u;
        sh[t] = v; __syncthreads();
        for (int off = 1; off < 1024; off <<= 1) {
            u32 xv = (t >= off) ? sh[t - off] : 0u;
            __syncthreads();
            sh[t] += xv;
            __syncthreads();
        }
        if (i < n) out[i] = carry + sh[t] - v;   // exclusive
        __syncthreads();
        if (t == 0) carry += sh[1023];
        __syncthreads();
    }
    if (t == 0) {
        if (total_out)   *total_out   = carry;
        if (total_f_out) *total_f_out = (float)carry;
    }
}

// ============================================================
// 2. Ranks from sorted node keys
// ============================================================
__global__ void ep_rank(const u64* keys, u32* rank_, unsigned npow, int n)
{
    unsigned p = blockIdx.x * blockDim.x + threadIdx.x;
    if (p >= npow) return;
    u32 idx = (u32)(keys[p] & 0xffffffffu);
    if ((int)idx < n) rank_[idx] = p;            // pads sort to the tail, idx>=n there
}

__global__ void ep_edge_rank(const long long* ei, const u32* rank_, u64* erank,
                             unsigned char* match, int E, int n)
{
    int e = blockIdx.x * blockDim.x + threadIdx.x;
    if (e >= E) return;
    u32 r = (u32)ei[e], c = (u32)ei[(size_t)E + e];
    erank[e] = (u64)rank_[r] * (u64)n + (u64)rank_[c];
    match[e] = 0;
}

// ============================================================
// 3. Locally-dominant maximal matching (fixed rounds + done flag)
// ============================================================
__global__ void ep_node_init(unsigned char* unmatched, unsigned char* hitv, u32* clusters, int n)
{
    int i = blockIdx.x * blockDim.x + threadIdx.x;
    if (i >= n) return;
    unmatched[i] = 1; hitv[i] = 0; clusters[i] = (u32)i;
}
__global__ void ep_ctrl_init(u32* ctrl)
{
    int t = threadIdx.x;
    if (t < 64) ctrl[t] = (t == 1) ? 1u : 0u;    // ctrl[1]=anyActive seed
}
__global__ void ep_match_reset(u64* nodemin, u32* ctrl, int n)
{
    int i = blockIdx.x * blockDim.x + threadIdx.x;
    if (i == 0 && blockIdx.x == 0) { if (ctrl[1] == 0u) ctrl[0] = 1u; ctrl[1] = 0u; }
    if (i < n) nodemin[i] = ~0ull;
}
__global__ void ep_scatter_min(const long long* ei, const u64* erank,
                               const unsigned char* unmatched, u64* nodemin, u32* ctrl, int E)
{
    if (ctrl[0]) return;
    int e = blockIdx.x * blockDim.x + threadIdx.x;
    if (e >= E) return;
    u32 r = (u32)ei[e], c = (u32)ei[(size_t)E + e];
    if (unmatched[r] && unmatched[c]) {
        u64 er = erank[e];
        atomicMin(&nodemin[r], er);              // global_atomic_min_u64
        atomicMin(&nodemin[c], er);
        ctrl[1] = 1u;
    }
}
__global__ void ep_dominant(const long long* ei, const u64* erank, const unsigned char* unmatched,
                            const u64* nodemin, unsigned char* match, unsigned char* hitv,
                            const u32* ctrl, int E)
{
    if (ctrl[0]) return;
    int e = blockIdx.x * blockDim.x + threadIdx.x;
    if (e >= E) return;
    u32 r = (u32)ei[e], c = (u32)ei[(size_t)E + e];
    if (!(unmatched[r] && unmatched[c])) return;
    u64 er = erank[e];
    if (er == nodemin[r] && er == nodemin[c]) { match[e] = 1; hitv[r] = 1; hitv[c] = 1; }
}
__global__ void ep_update(unsigned char* unmatched, unsigned char* hitv, const u32* ctrl, int n)
{
    if (ctrl[0]) return;
    int i = blockIdx.x * blockDim.x + threadIdx.x;
    if (i >= n) return;
    if (hitv[i]) { unmatched[i] = 0; hitv[i] = 0; }
}

// ============================================================
// 4. Cluster assignment + relabel (sorted-unique semantics)
// ============================================================
__global__ void ep_cluster_assign(const long long* ei, const unsigned char* match, u32* clusters, int E)
{
    int e = blockIdx.x * blockDim.x + threadIdx.x;
    if (e >= E || !match[e]) return;
    clusters[(u32)ei[(size_t)E + e]] = (u32)ei[e];
}
__global__ void ep_flag_reps(const u32* clusters, u32* flags, int n)
{
    int i = blockIdx.x * blockDim.x + threadIdx.x;
    if (i >= n) return;
    flags[clusters[i]] = 1u;
}
__global__ void ep_cfin(const u32* clusters, const u32* newid, u32* cfin, float* o_clu, int n)
{
    int i = blockIdx.x * blockDim.x + threadIdx.x;
    if (i >= n) return;
    u32 cid = newid[clusters[i]];
    cfin[i] = cid;
    o_clu[i] = (float)cid;
}

// ============================================================
// 5. x_pool = segment_sum(x, cluster)
// ============================================================
__global__ void ep_xpool(const float* __restrict__ x, const u32* __restrict__ cfin,
                         float* __restrict__ xpool, long long total, int d)
{
    long long t = (long long)blockIdx.x * blockDim.x + threadIdx.x;
    if (t >= total) return;
    int i = (int)(t / d), f = (int)(t % d);
    atomicAdd(&xpool[(size_t)cfin[i] * d + f], x[t]);
}

// ============================================================
// 6. Pooled-edge coalesce: sorted unique keys + segment-sum
// ============================================================
__global__ void ep_ekey(const long long* ei, const u32* cfin, u64* ekeys, int E, int n)
{
    int e = blockIdx.x * blockDim.x + threadIdx.x;
    if (e >= E) return;
    u32 r = (u32)ei[e], c = (u32)ei[(size_t)E + e];
    ekeys[e] = (u64)cfin[r] * (u64)n + (u64)cfin[c];
}
__global__ void ep_head_count(const u64* keys, u32* blocksums, unsigned npow)
{
    __shared__ u32 cnt;
    if (threadIdx.x == 0) cnt = 0;
    __syncthreads();
    unsigned p = blockIdx.x * 1024u + threadIdx.x;
    if (p < npow) {
        u64 k = keys[p];
        if (k != ~0ull && (p == 0 || keys[p - 1] != k)) atomicAdd(&cnt, 1u);
    }
    __syncthreads();
    if (threadIdx.x == 0) blocksums[blockIdx.x] = cnt;
}
__global__ void ep_compact_uk(const u64* keys, const u32* blockoff, u64* uk, unsigned npow)
{
    __shared__ u32 sh[1024];
    int t = threadIdx.x;
    unsigned p = blockIdx.x * 1024u + t;
    u64 k = (p < npow) ? keys[p] : ~0ull;
    u32 head = (p < npow) && (k != ~0ull) && (p == 0 || keys[p - 1] != k);
    sh[t] = head; __syncthreads();
    for (int off = 1; off < 1024; off <<= 1) {
        u32 xv = (t >= off) ? sh[t - off] : 0u;
        __syncthreads();
        sh[t] += xv;
        __syncthreads();
    }
    if (head) uk[blockoff[blockIdx.x] + sh[t] - 1] = k;
}
__global__ void ep_uk_tail(u64* uk, const u32* Uptr, int m, u64 inf)
{
    int s = blockIdx.x * blockDim.x + threadIdx.x;
    if (s < m && (u32)s >= *Uptr) uk[s] = inf;
}
__global__ void ep_valsum(const long long* ei, const float* val, const u32* cfin,
                          const u64* __restrict__ uk, float* vsum, int E, int n)
{
    int e = blockIdx.x * blockDim.x + threadIdx.x;
    if (e >= E) return;
    u32 r = (u32)ei[e], c = (u32)ei[(size_t)E + e];
    u64 key = (u64)cfin[r] * (u64)n + (u64)cfin[c];
    int lo = 0, hi = E;                      // uk sorted asc, tail = INF > key
    while (lo < hi) { int mid = (lo + hi) >> 1; if (uk[mid] < key) lo = mid + 1; else hi = mid; }
    atomicAdd(&vsum[lo], val[e]);
}
__global__ void ep_edge_out(const u64* uk, const float* vsum,
                            float* rowf, float* colf, float* valf, int E, int n, u64 inf)
{
    int s = blockIdx.x * blockDim.x + threadIdx.x;
    if (s >= E) return;
    u64 k = uk[s];
    u64 r = k / (u64)n, c = k % (u64)n;
    bool keep = (k != inf) && (r != c);
    rowf[s] = keep ? (float)r : (float)n;
    colf[s] = keep ? (float)c : (float)n;
    valf[s] = keep ? vsum[s] : 0.0f;
}
__global__ void ep_match_out(const unsigned char* match, float* o, int E)
{
    int e = blockIdx.x * blockDim.x + threadIdx.x;
    if (e < E) o[e] = match[e] ? 1.0f : 0.0f;
}

// ============================================================
// Host orchestration
// ============================================================
static inline int cdiv_i(long long a, long long b) { return (int)((a + b - 1) / b); }

static void bitonic_sort_u64(u64* a, unsigned npow, hipStream_t stream)
{
    // npow must be >= 1024 and a power of two
    unsigned nb = npow / 1024u;
    ep_bitonic_local<<<nb, 512, 0, stream>>>(a);                       // k = 2..1024
    for (unsigned k = 2048; k <= npow; k <<= 1) {
        for (unsigned j = k >> 1; j >= 1024; j >>= 1)
            ep_bitonic<<<cdiv_i(npow, TPB), TPB, 0, stream>>>(a, npow, j, k);
        ep_bitonic_fused<<<nb, 512, 0, stream>>>(a, k);                // j = 512..1 in LDS
    }
}

extern "C" void kernel_launch(void* const* d_in, const int* in_sizes, int n_in,
                              void* d_out, int out_size, void* d_ws, size_t ws_size,
                              hipStream_t stream)
{
    (void)n_in; (void)out_size; (void)ws_size;
    const float*     x    = (const float*)d_in[0];
    const long long* ei   = (const long long*)d_in[1];   // int64 edge_index (2,E)
    const float*     val  = (const float*)d_in[2];
    const float*     w    = (const float*)d_in[3];
    const float*     bias = (const float*)d_in[4];
    const int d = in_sizes[3];                 // 128
    const int n = in_sizes[0] / d;             // 100000
    const int E = in_sizes[2];                 // 3200000

    unsigned NPAD = 1024; while ((int)NPAD < n) NPAD <<= 1;
    unsigned EPAD = 1024; while ((int)EPAD < E) EPAD <<= 1;
    const unsigned nbE = EPAD / 1024u;
    const u64 INFKEY = (u64)n * (u64)n;

    // ---- d_out layout (float32, reference return order) ----
    float* o_xpool = (float*)d_out;                        // n*d
    float* o_row   = o_xpool + (size_t)n * d;              // E
    float* o_col   = o_row + E;                            // E
    float* o_val   = o_col + E;                            // E
    float* o_clu   = o_val + E;                            // n
    float* o_mat   = o_clu + n;                            // E
    float* o_sco   = o_mat + E;                            // n
    float* o_c     = o_sco + n;                            // 1

    // ---- workspace layout (~105 MB) ----
    char* p = (char*)d_ws;
    auto alloc = [&](size_t bytes) -> void* {
        void* r = (void*)p; p += (bytes + 255) & ~(size_t)255; return r;
    };
    u64* nodekeys = (u64*)alloc((size_t)NPAD * 8);
    u64* erank    = (u64*)alloc((size_t)E * 8);
    u64* nodemin  = (u64*)alloc((size_t)n * 8);
    u64* ekeys    = (u64*)alloc((size_t)EPAD * 8);
    u64* uk       = (u64*)alloc((size_t)E * 8);
    u32* rank_    = (u32*)alloc((size_t)n * 4);
    u32* clusters = (u32*)alloc((size_t)n * 4);
    u32* newid    = (u32*)alloc((size_t)n * 4);
    u32* cfin     = (u32*)alloc((size_t)n * 4);
    u32* flags    = (u32*)alloc((size_t)n * 4);
    float* vsum   = (float*)alloc((size_t)E * 4);
    u32* bsum     = (u32*)alloc((size_t)nbE * 4);
    u32* boff     = (u32*)alloc((size_t)nbE * 4);
    unsigned char* match     = (unsigned char*)alloc((size_t)E);
    unsigned char* unmatched = (unsigned char*)alloc((size_t)n);
    unsigned char* hitv      = (unsigned char*)alloc((size_t)n);
    u32* ctrl = (u32*)alloc(64 * 4);

    // ---- 1. scores (WMMA) + sortable node keys ----
    {
        long long waves = (n + 15) / 16;
        ep_score_wmma<<<cdiv_i(waves * 32, TPB), TPB, 0, stream>>>(x, w, bias, o_sco, nodekeys, n, d);
    }
    if (NPAD > (unsigned)n)
        ep_pad_u64<<<cdiv_i((long long)NPAD - n, TPB), TPB, 0, stream>>>(nodekeys, n, NPAD);

    // ---- 2. LDS-fused bitonic sort of node keys, derive ranks ----
    bitonic_sort_u64(nodekeys, NPAD, stream);
    ep_rank<<<cdiv_i(NPAD, TPB), TPB, 0, stream>>>(nodekeys, rank_, NPAD, n);
    ep_edge_rank<<<cdiv_i(E, TPB), TPB, 0, stream>>>(ei, rank_, erank, match, E, n);

    // ---- 3. matching loop (fixed rounds, device early-out) ----
    ep_node_init<<<cdiv_i(n, TPB), TPB, 0, stream>>>(unmatched, hitv, clusters, n);
    ep_ctrl_init<<<1, 64, 0, stream>>>(ctrl);
    for (int it = 0; it < 64; ++it) {
        ep_match_reset<<<cdiv_i(n, TPB), TPB, 0, stream>>>(nodemin, ctrl, n);
        ep_scatter_min<<<cdiv_i(E, TPB), TPB, 0, stream>>>(ei, erank, unmatched, nodemin, ctrl, E);
        ep_dominant<<<cdiv_i(E, TPB), TPB, 0, stream>>>(ei, erank, unmatched, nodemin, match, hitv, ctrl, E);
        ep_update<<<cdiv_i(n, TPB), TPB, 0, stream>>>(unmatched, hitv, ctrl, n);
    }

    // ---- 4. clusters + relabel (sorted unique) ----
    ep_cluster_assign<<<cdiv_i(E, TPB), TPB, 0, stream>>>(ei, match, clusters, E);
    ep_zero_u32<<<cdiv_i(n, TPB), TPB, 0, stream>>>(flags, n);
    ep_flag_reps<<<cdiv_i(n, TPB), TPB, 0, stream>>>(clusters, flags, n);
    ep_scan_u32<<<1, 1024, 0, stream>>>(flags, newid, n, &ctrl[4], o_c);
    ep_cfin<<<cdiv_i(n, TPB), TPB, 0, stream>>>(clusters, newid, cfin, o_clu, n);

    // ---- 5. x_pool ----
    ep_zero_f32<<<cdiv_i((long long)n * d, TPB), TPB, 0, stream>>>(o_xpool, (long long)n * d);
    ep_xpool<<<cdiv_i((long long)n * d, TPB), TPB, 0, stream>>>(x, cfin, o_xpool, (long long)n * d, d);

    // ---- 6. pooled-edge coalesce ----
    ep_ekey<<<cdiv_i(E, TPB), TPB, 0, stream>>>(ei, cfin, ekeys, E, n);
    if (EPAD > (unsigned)E)
        ep_pad_u64<<<cdiv_i((long long)EPAD - E, TPB), TPB, 0, stream>>>(ekeys, E, EPAD);
    bitonic_sort_u64(ekeys, EPAD, stream);
    ep_head_count<<<nbE, 1024, 0, stream>>>(ekeys, bsum, EPAD);
    ep_scan_u32<<<1, 1024, 0, stream>>>(bsum, boff, (int)nbE, &ctrl[5], nullptr);
    ep_compact_uk<<<nbE, 1024, 0, stream>>>(ekeys, boff, uk, EPAD);
    ep_uk_tail<<<cdiv_i(E, TPB), TPB, 0, stream>>>(uk, &ctrl[5], E, INFKEY);
    ep_zero_f32<<<cdiv_i(E, TPB), TPB, 0, stream>>>(vsum, E);
    ep_valsum<<<cdiv_i(E, TPB), TPB, 0, stream>>>(ei, val, cfin, uk, vsum, E, n);
    ep_edge_out<<<cdiv_i(E, TPB), TPB, 0, stream>>>(uk, vsum, o_row, o_col, o_val, E, n, INFKEY);
    ep_match_out<<<cdiv_i(E, TPB), TPB, 0, stream>>>(match, o_mat, E);
}
